// VectorQuantizer_70935679861413
// MI455X (gfx1250) — matrix operand order repaired and checked by reading, not compiled, not run on hardware
//
#include <hip/hip_runtime.h>
#include <hip/hip_bf16.h>

typedef __attribute__((ext_vector_type(16))) _Float16 v16h;
typedef __attribute__((ext_vector_type(8)))  _Float16 v8h;
typedef __attribute__((ext_vector_type(8)))  float    v8f;

#define N_ROWS   32768
#define N_EMB    8192
#define K_DIM    256
#define ROWS_PER_WAVE   32
#define WAVES_PER_BLOCK 4
#define BLOCK_THREADS   (WAVES_PER_BLOCK * 32)              // 128
#define ROWS_PER_BLOCK  (ROWS_PER_WAVE * WAVES_PER_BLOCK)   // 128
#define NUM_BLOCKS      (N_ROWS / ROWS_PER_BLOCK)           // 256
#define COLTILE         32
#define NUM_COLTILES    (N_EMB / COLTILE)                   // 256
#define LDSB_STRIDE     264   // 256 + 8 f16 pad (rows stay 16B aligned, banks spread)

// d_out layout (floats)
#define LOSS_OFF (N_ROWS * K_DIM)       // 8388608
#define IDX_OFF  (LOSS_OFF + 1)

// workspace layout (bytes)
#define WS_ESQ_OFF  (N_EMB * K_DIM * 2)         // after embH (4 MB)
#define WS_LOSS_OFF (WS_ESQ_OFF + N_EMB * 4)

// ---------------------------------------------------------------------------
// Kernel 1: embedding f32 -> f16 copy + ||e||^2 per row; zero loss accumulator
// ---------------------------------------------------------------------------
__global__ void vq_prep(const float* __restrict__ emb,
                        _Float16* __restrict__ embH,
                        float* __restrict__ esq,
                        float* __restrict__ lossAcc) {
  const int lane = threadIdx.x & 31;
  const int row  = blockIdx.x * 8 + (threadIdx.x >> 5);
  const float4* src = (const float4*)(emb + (size_t)row * K_DIM);
  float4 a = src[lane * 2 + 0];
  float4 b = src[lane * 2 + 1];
  v8h h;
  h[0] = (_Float16)a.x; h[1] = (_Float16)a.y; h[2] = (_Float16)a.z; h[3] = (_Float16)a.w;
  h[4] = (_Float16)b.x; h[5] = (_Float16)b.y; h[6] = (_Float16)b.z; h[7] = (_Float16)b.w;
  *(v8h*)(embH + (size_t)row * K_DIM + lane * 8) = h;
  float s = a.x*a.x + a.y*a.y + a.z*a.z + a.w*a.w
          + b.x*b.x + b.y*b.y + b.z*b.z + b.w*b.w;
#pragma unroll
  for (int d = 16; d >= 1; d >>= 1) s += __shfl_xor(s, d, 32);
  if (lane == 0) esq[row] = s;
  if (blockIdx.x == 0 && threadIdx.x == 0) *lossAcc = 0.0f;
}

// ---------------------------------------------------------------------------
// Kernel 2: fused distance-GEMM (WMMA f16->f32) + argmin + gather + loss.
// B tiles staged via global_load_async_to_lds_b128, double-buffered LDS,
// one barrier per tile.
// ---------------------------------------------------------------------------
__global__ void __launch_bounds__(BLOCK_THREADS)
vq_main(const float* __restrict__ Z,
        const float* __restrict__ emb,
        const _Float16* __restrict__ embH,
        const float* __restrict__ esq,
        float* __restrict__ out,
        float* __restrict__ lossAcc) {

  __shared__ __align__(16) _Float16 ldsB[2][COLTILE][LDSB_STRIDE];
  __shared__ int ldsIdx[ROWS_PER_BLOCK];

  const int tid  = threadIdx.x;
  const int lane = tid & 31;
  const int wave = tid >> 5;
  const int lrow = lane & 15;     // M (A) or N (B/C) index inside 16-tile
  const int half = lane >> 4;     // K-half selector for A/B fragments
  const int waveRow0 = blockIdx.x * ROWS_PER_BLOCK + wave * ROWS_PER_WAVE;

  // ---- A operand resident in registers: 32 rows x 256 K, f32 -> f16 ----
  // 16-bit A 16x32 layout: lanes 0-15 hold K = kk*32 + {0..7, 16..23},
  // lanes 16-31 hold K = kk*32 + {8..15, 24..31}.
  v16h aReg[2][8];
#pragma unroll
  for (int rg = 0; rg < 2; ++rg) {
    const float4* zr =
        (const float4*)(Z + (size_t)(waveRow0 + rg * 16 + lrow) * K_DIM);
#pragma unroll
    for (int kk = 0; kk < 8; ++kk) {
      const int q = kk * 8 + half * 2;  // float4 index of K = kk*32 + half*8
      float4 a0 = zr[q], a1 = zr[q + 1], a2 = zr[q + 4], a3 = zr[q + 5];
      v16h f;
      f[0]  = (_Float16)a0.x; f[1]  = (_Float16)a0.y;
      f[2]  = (_Float16)a0.z; f[3]  = (_Float16)a0.w;
      f[4]  = (_Float16)a1.x; f[5]  = (_Float16)a1.y;
      f[6]  = (_Float16)a1.z; f[7]  = (_Float16)a1.w;
      f[8]  = (_Float16)a2.x; f[9]  = (_Float16)a2.y;
      f[10] = (_Float16)a2.z; f[11] = (_Float16)a2.w;
      f[12] = (_Float16)a3.x; f[13] = (_Float16)a3.y;
      f[14] = (_Float16)a3.z; f[15] = (_Float16)a3.w;
      aReg[rg][kk] = f;
    }
  }

  float mind[2][8];
  int   mini[2][8];
#pragma unroll
  for (int rg = 0; rg < 2; ++rg)
#pragma unroll
    for (int r = 0; r < 8; ++r) { mind[rg][r] = 3.4e38f; mini[rg][r] = 0; }

  // ---- prologue: async-stage tile 0 into buffer 0 (8 b128 per thread-loop) ----
#pragma unroll
  for (int i = 0; i < 8; ++i) {
    const int t = tid + i * BLOCK_THREADS;
    const int el = t >> 5, seg = t & 31;               // 32 x 16B per emb row
    unsigned int  ldsOff = (unsigned int)(size_t)&ldsB[0][el][seg * 8];
    unsigned long long ga =
        (unsigned long long)(size_t)(embH + (size_t)el * K_DIM + seg * 8);
    asm volatile("global_load_async_to_lds_b128 %0, %1, off"
                 :: "v"(ldsOff), "v"(ga) : "memory");
  }

  for (int ct = 0; ct < NUM_COLTILES; ++ct) {
    // wait for this wave's async stores into LDS, then block-wide sync
    asm volatile("s_wait_asynccnt 0x0" ::: "memory");
    __syncthreads();

    // issue async stage of tile ct+1 into the other buffer (uniform branch)
    if (ct + 1 < NUM_COLTILES) {
      const int nb = (ct + 1) & 1;
#pragma unroll
      for (int i = 0; i < 8; ++i) {
        const int t = tid + i * BLOCK_THREADS;
        const int el = t >> 5, seg = t & 31;
        unsigned int ldsOff = (unsigned int)(size_t)&ldsB[nb][el][seg * 8];
        unsigned long long ga = (unsigned long long)(size_t)(
            embH + ((size_t)(ct + 1) * COLTILE + el) * K_DIM + seg * 8);
        asm volatile("global_load_async_to_lds_b128 %0, %1, off"
                     :: "v"(ldsOff), "v"(ga) : "memory");
      }
    }

    const int cb = ct & 1;
    v8f acc[2][2];
#pragma unroll
    for (int rg = 0; rg < 2; ++rg) {
      v8f z8 = {};
      acc[rg][0] = z8; acc[rg][1] = z8;
    }

#pragma unroll
    for (int kk = 0; kk < 8; ++kk) {
      // B 32x16 layout: lane N = L%16, K = kk*32 + (L/16)*8 + {0..7, 16..23}
      const int k = kk * 32 + half * 8;
      v16h b0, b1;
      {
        const v8h* p0 = (const v8h*)&ldsB[cb][lrow][k];
        v8h lo = p0[0], hi = p0[2];
#pragma unroll
        for (int i = 0; i < 8; ++i) { b0[i] = lo[i]; b0[8 + i] = hi[i]; }
        const v8h* p1 = (const v8h*)&ldsB[cb][16 + lrow][k];
        lo = p1[0]; hi = p1[2];
#pragma unroll
        for (int i = 0; i < 8; ++i) { b1[i] = lo[i]; b1[8 + i] = hi[i]; }
      }
#pragma unroll
      for (int rg = 0; rg < 2; ++rg) {
        acc[rg][0] = __builtin_amdgcn_wmma_f32_16x16x32_f16(
            false, aReg[rg][kk], false, b0, (short)0, acc[rg][0], false, false);
        acc[rg][1] = __builtin_amdgcn_wmma_f32_16x16x32_f16(
            false, aReg[rg][kk], false, b1, (short)0, acc[rg][1], false, false);
      }
    }

    // ---- epilogue: d = ||e||^2 - 2 z.e ; running argmin per C row ----
    const int   e0  = ct * COLTILE + lrow;
    const float es0 = esq[e0];
    const float es1 = esq[e0 + 16];
#pragma unroll
    for (int rg = 0; rg < 2; ++rg)
#pragma unroll
      for (int r = 0; r < 8; ++r) {
        float d0 = __builtin_fmaf(-2.0f, acc[rg][0][r], es0);
        if (d0 < mind[rg][r]) { mind[rg][r] = d0; mini[rg][r] = e0; }
        float d1 = __builtin_fmaf(-2.0f, acc[rg][1][r], es1);
        if (d1 < mind[rg][r]) { mind[rg][r] = d1; mini[rg][r] = e0 + 16; }
      }
  }

  // ---- cross-lane argmin: C row (rg,r) lives in lanes 0-15 (M=r) and
  //      lanes 16-31 (M=8+r); reduce within each 16-lane half ----
  __syncthreads();
#pragma unroll
  for (int rg = 0; rg < 2; ++rg)
#pragma unroll
    for (int r = 0; r < 8; ++r) {
      float m  = mind[rg][r];
      int   mi = mini[rg][r];
#pragma unroll
      for (int d = 1; d < 16; d <<= 1) {
        float om = __shfl_xor(m, d, 32);
        int   oi = __shfl_xor(mi, d, 32);
        if (om < m || (om == m && oi < mi)) { m = om; mi = oi; }
      }
      if (lrow == 0) {
        const int rloc = wave * ROWS_PER_WAVE + rg * 16 + half * 8 + r;
        ldsIdx[rloc] = mi;
        out[IDX_OFF + blockIdx.x * ROWS_PER_BLOCK + rloc] = (float)mi;
      }
    }
  __syncthreads();

  // ---- gather z_q = embedding[idx] (f32), write out, accumulate loss ----
  const float4* embV = (const float4*)emb;
  const float4* zV   = (const float4*)(Z + (size_t)blockIdx.x * ROWS_PER_BLOCK * K_DIM);
  float4* outV = (float4*)(out + (size_t)blockIdx.x * ROWS_PER_BLOCK * K_DIM);
  float ls = 0.0f;
  for (int i = tid; i < ROWS_PER_BLOCK * (K_DIM / 4); i += BLOCK_THREADS) {
    const int r = i >> 6;                     // 64 float4 per row
    const int e = ldsIdx[r];
    float4 ev = embV[(size_t)e * (K_DIM / 4) + (i & 63)];
    float4 zv = zV[i];
    outV[i] = ev;
    const float dx = ev.x - zv.x, dy = ev.y - zv.y;
    const float dz = ev.z - zv.z, dw = ev.w - zv.w;
    ls += dx * dx + dy * dy + dz * dz + dw * dw;
  }
#pragma unroll
  for (int d = 16; d >= 1; d >>= 1) ls += __shfl_xor(ls, d, 32);
  if (lane == 0) atomicAdd(lossAcc, ls);
}

// ---------------------------------------------------------------------------
// Kernel 3: loss = 1.25 * sum(diff^2) / numel
// ---------------------------------------------------------------------------
__global__ void vq_finish(const float* __restrict__ lossAcc, float* __restrict__ out) {
  out[LOSS_OFF] = 1.25f * (*lossAcc) / (float)(N_ROWS * K_DIM);
}

extern "C" void kernel_launch(void* const* d_in, const int* in_sizes, int n_in,
                              void* d_out, int out_size, void* d_ws, size_t ws_size,
                              hipStream_t stream) {
  (void)in_sizes; (void)n_in; (void)out_size; (void)ws_size;
  const float* z   = (const float*)d_in[0];   // (8,256,64,64) f32 == (32768,256)
  const float* emb = (const float*)d_in[1];   // (8192,256) f32
  float* out = (float*)d_out;

  char* ws = (char*)d_ws;
  _Float16* embH    = (_Float16*)ws;                // 4 MB f16 embedding copy
  float*    esq     = (float*)(ws + WS_ESQ_OFF);    // 8192 f32 row norms
  float*    lossAcc = (float*)(ws + WS_LOSS_OFF);   // 1 f32 accumulator

  vq_prep<<<N_EMB / 8, 256, 0, stream>>>(emb, embH, esq, lossAcc);
  vq_main<<<NUM_BLOCKS, BLOCK_THREADS, 0, stream>>>(z, emb, embH, esq,
                                                    out, lossAcc);
  vq_finish<<<1, 1, 0, stream>>>(lossAcc, out);
}
